// CARAFE_MSGConv_82532091560404
// MI455X (gfx1250) — matrix operand order repaired and checked by reading, not compiled, not run on hardware
//
#include <hip/hip_runtime.h>
#include <hip/hip_bf16.h>
#include <math.h>

// ---------------------------------------------------------------------------
// CARAFE + MSGConv for MI455X (gfx1250, wave32, WMMA).
// 1x1 convs -> bf16 WMMA GEMMs (v_wmma_f32_16x16x32_bf16, fp32 accum).
// Depthwise convs + CARAFE softmax/gather -> bandwidth-oriented VALU kernels.
// Whole problem is HBM-bound (~30 MB traffic @ 23.3 TB/s); bf16 matmul
// inputs cost nothing in time and free the VALU; K<=128 keeps fp32 accum
// plenty accurate.
// This revision: branchless LDS staging (clamp+select), vectorized bf16 LDS
// stores (b64/b128), double-buffered K-loop (one barrier per step).
// ---------------------------------------------------------------------------

typedef __bf16 bf16;
typedef __attribute__((ext_vector_type(16))) __bf16 v16bf;
typedef __attribute__((ext_vector_type(8)))  __bf16 v8bf;
typedef __attribute__((ext_vector_type(4)))  __bf16 v4bf;
typedef __attribute__((ext_vector_type(8)))  float  v8f;

#define HW   4096   // 64*64 spatial
#define NPB  4096   // GEMM N per batch
#define LSTR 40     // LDS row stride in bf16 elems (80B: 16B-aligned, bank-skewed)

__device__ __forceinline__ float silu_f(float v) { return v / (1.0f + __expf(-v)); }

// ---------------------------------------------------------------------------
// GEMM: C[M x 4096] = A[M x K] * B[K x 4096]  (per batch, grid.z = batch)
// then C = silu(C * scale[m] + bias[m]).
// Block: 256 thr = 8 waves; WG tile 32x64; one 16x16 WMMA tile per wave.
// K padded to mult of 32 via zero-fill (handles K=100); M=50 handled by
// zero rows + store guard. Double-buffered LDS staging.
// ---------------------------------------------------------------------------
template<int M, int K>
__global__ __launch_bounds__(256)
void gemm_bn_silu(const float* __restrict__ A,
                  const float* __restrict__ B,
                  const float* __restrict__ scale,
                  const float* __restrict__ bias,
                  float* __restrict__ C,
                  int bStrideB, int bStrideC)
{
  constexpr int Kp = (K + 31) & ~31;
  constexpr bool ALIGNED = (M % 32 == 0) && (K % 32 == 0);
  __shared__ __align__(16) bf16 As[2][32 * LSTR];   // [m][k]
  __shared__ __align__(16) bf16 Bs[2][64 * LSTR];   // [n][k] (transposed tile)

  const int tid  = threadIdx.x;
  const int wave = tid >> 5;
  const int lane = tid & 31;
  const int half = lane >> 4;
  const int l16  = lane & 15;
  const int mi = wave >> 2;            // 0..1 -> M sub-tile
  const int ni = wave & 3;             // 0..3 -> N sub-tile
  const int m0 = blockIdx.y * 32;
  const int n0 = blockIdx.x * 64;
  const int b  = blockIdx.z;

  const float* Bb = B + (size_t)b * bStrideB;
  float*       Cb = C + (size_t)b * bStrideC;

  // staging ownership: A -> thread owns (m = tid>>3, k = (tid&7)*4 .. +3)
  //                    B -> thread owns (n = tid&63, k = (tid>>6)*8 .. +7)
  const int am = tid >> 3;
  const int ak = (tid & 7) * 4;
  const int bn = tid & 63;
  const int bk = (tid >> 6) * 8;

  auto stage = [&](int k0, int buf) {
    // ---- A tile 32x32: 4 contiguous k per thread, one b64 LDS store ----
    float av[4];
#pragma unroll
    for (int j = 0; j < 4; ++j) {
      const int gm = m0 + am, gk = k0 + ak + j;
      if constexpr (ALIGNED) {
        av[j] = A[gm * K + gk];
      } else {
        const int gmc = gm < M ? gm : M - 1;       // clamp: load always legal
        const int gkc = gk < K ? gk : K - 1;
        float v = A[gmc * K + gkc];
        av[j] = (gm < M && gk < K) ? v : 0.0f;     // branchless v_cndmask
      }
    }
    v4bf ap;
#pragma unroll
    for (int j = 0; j < 4; ++j) ap[j] = (bf16)av[j];
    *reinterpret_cast<v4bf*>(&As[buf][am * LSTR + ak]) = ap;

    // ---- B tile 32(K)x64(N): 8 contiguous k per thread, one b128 store ----
    float bv[8];
#pragma unroll
    for (int j = 0; j < 8; ++j) {
      const int gk = k0 + bk + j;
      if constexpr (K % 32 == 0) {
        bv[j] = Bb[(size_t)gk * NPB + n0 + bn];
      } else {
        const int gkc = gk < K ? gk : K - 1;
        float v = Bb[(size_t)gkc * NPB + n0 + bn];
        bv[j] = (gk < K) ? v : 0.0f;
      }
    }
    v8bf bp;
#pragma unroll
    for (int j = 0; j < 8; ++j) bp[j] = (bf16)bv[j];
    *reinterpret_cast<v8bf*>(&Bs[buf][bn * LSTR + bk]) = bp;
  };

  v8f acc = {};
  stage(0, 0);

  int buf = 0;
#pragma unroll
  for (int k0 = 0; k0 < Kp; k0 += 32) {
    __syncthreads();
    if (k0 + 32 < Kp) stage(k0 + 32, buf ^ 1);   // overlap next-tile loads

    // A frag (16x32 16-bit): elems 0-7 <- K=half*8+0..7, 8-15 <- K=16+half*8+0..7
    const int mr = mi * 16 + l16;
    v8bf alo = *reinterpret_cast<const v8bf*>(&As[buf][mr * LSTR + half * 8]);
    v8bf ahi = *reinterpret_cast<const v8bf*>(&As[buf][mr * LSTR + 16 + half * 8]);
    // B frag (32x16): elems 0-7 <- K=half*8+0..7, 8-15 <- K=16+half*8+0..7
    const int nr = ni * 16 + l16;
    v8bf blo = *reinterpret_cast<const v8bf*>(&Bs[buf][nr * LSTR + half * 8]);
    v8bf bhi = *reinterpret_cast<const v8bf*>(&Bs[buf][nr * LSTR + 16 + half * 8]);
    v16bf af, bfr;
#pragma unroll
    for (int i = 0; i < 8; ++i) {
      af[i] = alo[i];  af[8 + i] = ahi[i];
      bfr[i] = blo[i]; bfr[8 + i] = bhi[i];
    }
    acc = __builtin_amdgcn_wmma_f32_16x16x32_bf16(
        /*neg_a=*/false, af, /*neg_b=*/false, bfr,
        /*c_mod=*/(short)0, acc, /*reuse_a=*/false, /*reuse_b=*/false);
    buf ^= 1;
  }

  // epilogue: C/D layout -> m = half*8 + r, n = lane&15
#pragma unroll
  for (int r = 0; r < 8; ++r) {
    const int m = m0 + mi * 16 + half * 8 + r;
    const int n = n0 + ni * 16 + l16;
    if ((M % 32 == 0) || (m < M)) {
      float v = acc[r] * scale[m] + bias[m];
      Cb[(size_t)m * NPB + n] = silu_f(v);
    }
  }
}

// ---------------------------------------------------------------------------
// Depthwise KxK SAME conv + BN + SiLU on 64x64 maps.
// ---------------------------------------------------------------------------
template<int KSZ>
__global__ void dw_bn_silu(const float* __restrict__ in,   // base of input chans
                           const float* __restrict__ w,    // [CH][KSZ*KSZ]
                           const float* __restrict__ scale,
                           const float* __restrict__ bias,
                           float* __restrict__ out,        // base of output chans
                           int CH, int inBS, int outBS)    // per-batch strides
{
  int tid = blockIdx.x * blockDim.x + threadIdx.x;
  int total = 2 * CH * HW;
  if (tid >= total) return;
  int x = tid & 63;
  int y = (tid >> 6) & 63;
  int c = (tid >> 12) % CH;
  int b = tid / (CH * HW);
  const float* ip = in + (size_t)b * inBS + (size_t)c * HW;
  const float* wp = w + c * KSZ * KSZ;
  constexpr int R = KSZ / 2;
  float acc = 0.f;
#pragma unroll
  for (int dy = 0; dy < KSZ; ++dy) {
    int yy = y + dy - R;
    if ((unsigned)yy >= 64u) continue;
#pragma unroll
    for (int dx = 0; dx < KSZ; ++dx) {
      int xx = x + dx - R;
      if ((unsigned)xx >= 64u) continue;
      acc += ip[yy * 64 + xx] * wp[dy * KSZ + dx];
    }
  }
  float v = acc * scale[c] + bias[c];
  out[(size_t)b * outBS + (size_t)c * HW + y * 64 + x] = silu_f(v);
}

// ---------------------------------------------------------------------------
// CARAFE: per-output-pixel softmax over 25 weights (pixel-shuffle phase picks
// channel k2*4 + (oy&1)*2 + (ox&1) of W0[b,100,64,64]), then 5x5 weighted
// gather over original-resolution X (src row = h+p-2, col = w+q-2, zero OOB).
// Block = 16x16 output tile x 32-channel group; 12x12 halo staged in LDS.
// ---------------------------------------------------------------------------
__global__ __launch_bounds__(256)
void carafe_gather(const float* __restrict__ X,    // [2,128,64,64]
                   const float* __restrict__ W0,   // [2,100,64,64]
                   float* __restrict__ out)        // [2,128,128,128]
{
  __shared__ float xs[32 * 156];                   // 32ch x 12 rows x 13(pad)
  const int tid = threadIdx.x;
  const int b   = blockIdx.z;
  const int c0  = blockIdx.y * 32;
  const int t   = blockIdx.x;                      // 0..63 -> 8x8 tiles
  const int ty = t >> 3, tx = t & 7;
  const int wol = tid & 15, hol = tid >> 4;
  const int oy = ty * 16 + hol, ox = tx * 16 + wol;
  const int h = oy >> 1, w = ox >> 1;
  const int off = (oy & 1) * 2 + (ox & 1);

  // softmax over the 25 kernel weights for this output pixel
  float wg[25];
  const float* Wp = W0 + ((size_t)b * 100 + off) * HW + h * 64 + w;
  float wmax = -3.4e38f;
#pragma unroll
  for (int j = 0; j < 25; ++j) { wg[j] = Wp[(size_t)j * 4 * HW]; wmax = fmaxf(wmax, wg[j]); }
  float sum = 0.f;
#pragma unroll
  for (int j = 0; j < 25; ++j) { wg[j] = __expf(wg[j] - wmax); sum += wg[j]; }
  float inv = 1.f / sum;
#pragma unroll
  for (int j = 0; j < 25; ++j) wg[j] *= inv;

  // stage 12x12 source halo for 32 channels (zero outside [0,64)^2)
  const int hs0 = ty * 8 - 2, ws0 = tx * 8 - 2;
#pragma unroll
  for (int i = 0; i < 18; ++i) {
    int li = tid + i * 256;                        // 0..4607 = 32*144
    int c = li / 144, rem = li - c * 144;
    int yy = rem / 12, xx = rem - yy * 12;
    int gy = hs0 + yy, gx = ws0 + xx;
    float v = 0.f;
    if ((unsigned)gy < 64u && (unsigned)gx < 64u)
      v = X[(((size_t)b * 128 + c0 + c) * 64 + gy) * 64 + gx];
    xs[c * 156 + yy * 13 + xx] = v;
  }
  __syncthreads();

  const int hl = hol >> 1, wl = wol >> 1;          // local src row/col for p=q=0
  float* op = out + (((size_t)b * 128 + c0) * 128 + oy) * 128 + ox;
#pragma unroll 4
  for (int c = 0; c < 32; ++c) {
    const float* xp = &xs[c * 156 + hl * 13 + wl];
    float acc = 0.f;
#pragma unroll
    for (int p = 0; p < 5; ++p)
#pragma unroll
      for (int q = 0; q < 5; ++q)
        acc += wg[p * 5 + q] * xp[p * 13 + q];
    op[(size_t)c * 128 * 128] = acc;
  }
}

// ---------------------------------------------------------------------------
extern "C" void kernel_launch(void* const* d_in, const int* in_sizes, int n_in,
                              void* d_out, int out_size, void* d_ws, size_t ws_size,
                              hipStream_t stream) {
  (void)in_sizes; (void)n_in; (void)out_size; (void)ws_size;
  const float* X    = (const float*)d_in[0];
  const float* c1w  = (const float*)d_in[1];
  const float* c1s  = (const float*)d_in[2];
  const float* c1b  = (const float*)d_in[3];
  const float* cd3w = (const float*)d_in[4];
  const float* cd3s = (const float*)d_in[5];
  const float* cd3b = (const float*)d_in[6];
  const float* cd5w = (const float*)d_in[7];
  const float* cd5s = (const float*)d_in[8];
  const float* cd5b = (const float*)d_in[9];
  const float* cpxw = (const float*)d_in[10];
  const float* cpxs = (const float*)d_in[11];
  const float* cpxb = (const float*)d_in[12];
  const float* e1w  = (const float*)d_in[13];
  const float* e1s  = (const float*)d_in[14];
  const float* e1b  = (const float*)d_in[15];
  const float* ed3w = (const float*)d_in[16];
  const float* ed3s = (const float*)d_in[17];
  const float* ed3b = (const float*)d_in[18];
  const float* ed5w = (const float*)d_in[19];
  const float* ed5s = (const float*)d_in[20];
  const float* ed5b = (const float*)d_in[21];
  const float* epxw = (const float*)d_in[22];
  const float* epxs = (const float*)d_in[23];
  const float* epxb = (const float*)d_in[24];

  // Workspace (floats). Layout (aliased; total ~7.1 MB):
  //   xcat  [2][64][4096]  @ 0          (x1 rows 0..31, x2 rows 32..63)
  //   ycomp [2][64][4096]  @ 524288
  //   ecat  [2][100][4096] @ 1048576    (x1e rows 0..49, x2e rows 50..99)
  //   W0    [2][100][4096] @ 0          (reuses xcat/ycomp, both dead by then)
  float* ws    = (float*)d_ws;
  float* xcat  = ws;
  float* ycomp = ws + (size_t)2 * 64 * HW;
  float* ecat  = ws + (size_t)4 * 64 * HW;
  float* W0    = ws;

  const dim3 blk(256);

  // 1) comp cv1: [32x128] x [128x4096] -> xcat rows 0..31
  gemm_bn_silu<32, 128><<<dim3(64, 1, 2), blk, 0, stream>>>(
      c1w, X, c1s, c1b, xcat, 128 * HW, 64 * HW);

  // 2) comp depthwise: dw3 on ch 0..15 -> rows 32..47; dw5 on ch 16..31 -> rows 48..63
  {
    int tot = 2 * 16 * HW;
    dw_bn_silu<3><<<(tot + 255) / 256, blk, 0, stream>>>(
        xcat, cd3w, cd3s, cd3b, xcat + (size_t)32 * HW, 16, 64 * HW, 64 * HW);
    dw_bn_silu<5><<<(tot + 255) / 256, blk, 0, stream>>>(
        xcat + (size_t)16 * HW, cd5w, cd5s, cd5b, xcat + (size_t)48 * HW, 16, 64 * HW, 64 * HW);
  }

  // 3) comp px: [64x64] x [64x4096] -> ycomp
  gemm_bn_silu<64, 64><<<dim3(64, 2, 2), blk, 0, stream>>>(
      cpxw, xcat, cpxs, cpxb, ycomp, 64 * HW, 64 * HW);

  // 4) enc cv1: [50x64] x [64x4096] -> ecat rows 0..49
  gemm_bn_silu<50, 64><<<dim3(64, 2, 2), blk, 0, stream>>>(
      e1w, ycomp, e1s, e1b, ecat, 64 * HW, 100 * HW);

  // 5) enc depthwise: dw3 on ch 0..24 -> rows 50..74; dw5 on ch 25..49 -> rows 75..99
  {
    int tot = 2 * 25 * HW;
    dw_bn_silu<3><<<(tot + 255) / 256, blk, 0, stream>>>(
        ecat, ed3w, ed3s, ed3b, ecat + (size_t)50 * HW, 25, 100 * HW, 100 * HW);
    dw_bn_silu<5><<<(tot + 255) / 256, blk, 0, stream>>>(
        ecat + (size_t)25 * HW, ed5w, ed5s, ed5b, ecat + (size_t)75 * HW, 25, 100 * HW, 100 * HW);
  }

  // 6) enc px: [100x100] x [100x4096] -> W0 (K zero-padded to 128 in-kernel)
  gemm_bn_silu<100, 100><<<dim3(64, 4, 2), blk, 0, stream>>>(
      epxw, ecat, epxs, epxb, W0, 100 * HW, 100 * HW);

  // 7) CARAFE: softmax + 5x5 gather -> out [2,128,128,128]
  carafe_gather<<<dim3(64, 4, 2), blk, 0, stream>>>(X, W0, (float*)d_out);
}
